// scaled_dot_product_attention_19086834663472
// MI455X (gfx1250) — compile-verified, matrix operand order checked by reading, not run on hardware
//
#include <hip/hip_runtime.h>

typedef __attribute__((ext_vector_type(16))) _Float16 v16h;
typedef __attribute__((ext_vector_type(8)))  _Float16 v8h;
typedef __attribute__((ext_vector_type(8)))  float    v8f;

#define BB 4
#define HH 16
#define SS 2048
#define DD 64

// 1/sqrt(64) * log2(e): exp2(scale*q.k) == exp((q.k)/8)
#define QSCALE   (0.125f * 1.44269504088896340736f)
#define MASKVAL  (-10000.0f * 1.44269504088896340736f)

__device__ __forceinline__ v8f wmma16x16x32(v16h a, v16h b, v8f c) {
  return __builtin_amdgcn_wmma_f32_16x16x32_f16(false, a, false, b, (short)0, c, false, false);
}

// Convert 8 consecutive scaled floats into halves dst[off .. off+7]
__device__ __forceinline__ void ld8(v16h& dst, int off, const float* __restrict__ p, float scale) {
  const float4 a = *reinterpret_cast<const float4*>(p);
  const float4 b = *reinterpret_cast<const float4*>(p + 4);
  dst[off + 0] = (_Float16)(a.x * scale);
  dst[off + 1] = (_Float16)(a.y * scale);
  dst[off + 2] = (_Float16)(a.z * scale);
  dst[off + 3] = (_Float16)(a.w * scale);
  dst[off + 4] = (_Float16)(b.x * scale);
  dst[off + 5] = (_Float16)(b.y * scale);
  dst[off + 6] = (_Float16)(b.z * scale);
  dst[off + 7] = (_Float16)(b.w * scale);
}

// Build a 16-half WMMA operand from two contiguous 16B runs (f16 in memory)
__device__ __forceinline__ v16h ldfrag(const _Float16* __restrict__ p0,
                                       const _Float16* __restrict__ p1) {
  union { v16h v; v8h h[2]; } u;
  u.h[0] = *reinterpret_cast<const v8h*>(p0);
  u.h[1] = *reinterpret_cast<const v8h*>(p1);
  return u.v;
}

// ---------------- pre-pass: K f32 -> f16 (row major) ----------------
__global__ __launch_bounds__(256, 1)
void cvt_f32_f16(const float* __restrict__ src, _Float16* __restrict__ dst) {
  const size_t i = ((size_t)blockIdx.x * blockDim.x + threadIdx.x) * 8;
  const float4 a = *reinterpret_cast<const float4*>(src + i);
  const float4 b = *reinterpret_cast<const float4*>(src + i + 4);
  v8h h;
  h[0] = (_Float16)a.x; h[1] = (_Float16)a.y; h[2] = (_Float16)a.z; h[3] = (_Float16)a.w;
  h[4] = (_Float16)b.x; h[5] = (_Float16)b.y; h[6] = (_Float16)b.z; h[7] = (_Float16)b.w;
  *reinterpret_cast<v8h*>(dst + i) = h;
}

// ---------------- pre-pass: V f32 [bh][s][d] -> f16 V^T [bh][d][s] ----------------
__global__ __launch_bounds__(256, 1)
void transpose_v_f16(const float* __restrict__ V, _Float16* __restrict__ Vt) {
  __shared__ _Float16 tile[DD][64 + 8];     // [d][s_local], padded
  const int bh   = blockIdx.x >> 5;         // / (SS/64)
  const int sblk = blockIdx.x & 31;
  const int r  = threadIdx.x >> 2;          // 0..63
  const int c0 = (threadIdx.x & 3) * 16;    // 0,16,32,48

  const float* __restrict__ Vp = V + ((size_t)bh * SS + (size_t)sblk * 64) * DD;
  #pragma unroll
  for (int j = 0; j < 4; ++j) {             // row s_local=r, d = c0+4j..+3
    const float4 x = *reinterpret_cast<const float4*>(Vp + (size_t)r * DD + c0 + j * 4);
    tile[c0 + j * 4 + 0][r] = (_Float16)x.x;
    tile[c0 + j * 4 + 1][r] = (_Float16)x.y;
    tile[c0 + j * 4 + 2][r] = (_Float16)x.z;
    tile[c0 + j * 4 + 3][r] = (_Float16)x.w;
  }
  __syncthreads();
  // write d-row r, s_local run [c0, c0+16)
  _Float16* __restrict__ out = Vt + ((size_t)bh * DD + r) * SS + (size_t)sblk * 64 + c0;
  v8h h0, h1;
  #pragma unroll
  for (int e = 0; e < 8; ++e) { h0[e] = tile[r][c0 + e]; h1[e] = tile[r][c0 + 8 + e]; }
  *reinterpret_cast<v8h*>(out)     = h0;
  *reinterpret_cast<v8h*>(out + 8) = h1;
}

// ---------------- main flash-attention kernel (f16 K / transposed f16 V) ----------------
// Each workgroup processes TWO mirrored 128-row q-chunks so all WGs do equal
// causal work. Each wave owns a 16-query tile per chunk.
__global__ __launch_bounds__(256, 1)
void fa_causal_fwd_h(const float* __restrict__ Q, const _Float16* __restrict__ Kh,
                     const _Float16* __restrict__ Vt, float* __restrict__ O) {
  const int lane = threadIdx.x & 31;
  const int wave = threadIdx.x >> 5;
  const int n = lane & 15;   // matrix column (query) index
  const int g = lane >> 4;   // half-wave group per ISA operand striping

  const int chunksPerSeq = SS / 128;                 // 16
  const int pairsPerSeq  = chunksPerSeq / 2;         // 8
  const int bh   = blockIdx.x / pairsPerSeq;
  const int pair = blockIdx.x % pairsPerSeq;

  const float*    __restrict__ Qp = Q  + (size_t)bh * SS * DD;
  const _Float16* __restrict__ Kp = Kh + (size_t)bh * SS * DD;
  const _Float16* __restrict__ Vp = Vt + (size_t)bh * DD * SS;   // [d][s]
  float*          __restrict__ Op = O  + (size_t)bh * SS * DD;

  for (int half = 0; half < 2; ++half) {
    const int qchunk = half ? (chunksPerSeq - 1 - pair) : pair;  // mirror pairing
    const int qbase  = qchunk * 128 + wave * 16;

    // Q^T B-operand, scale folded (includes log2e)
    v16h qb[2];
    {
      const float* qrow = Qp + (size_t)(qbase + n) * DD;
      #pragma unroll
      for (int c = 0; c < 2; ++c) {
        const int d0 = 8 * g + 32 * c;
        ld8(qb[c], 0, qrow + d0,      QSCALE);
        ld8(qb[c], 8, qrow + d0 + 16, QSCALE);
      }
    }

    v8f oacc[4];
    #pragma unroll
    for (int t = 0; t < 4; ++t) {
      #pragma unroll
      for (int r = 0; r < 8; ++r) oacc[t][r] = 0.0f;
    }
    float m_i = -3.0e38f;
    float l_i = 0.0f;

    const int kend = qbase + 16;
    for (int kb = 0; kb < kend; kb += 32) {
      if (kb + 32 < kend)
        __builtin_prefetch(Kp + (size_t)(kb + 32) * DD + (size_t)lane * 8, 0, 1);

      // Issue V^T fragment loads first: independent of the scores, so their
      // latency hides under QK^T WMMA + softmax.
      v16h va[4];
      #pragma unroll
      for (int t = 0; t < 4; ++t) {
        const _Float16* vrow = Vp + (size_t)(16 * t + n) * SS + kb + 8 * g;
        va[t] = ldfrag(vrow, vrow + 16);
      }

      // S^T = K_block * Q^T  (scores already in log2 domain via QSCALE)
      v8f s0, s1;
      #pragma unroll
      for (int r = 0; r < 8; ++r) { s0[r] = 0.0f; s1[r] = 0.0f; }
      #pragma unroll
      for (int c = 0; c < 2; ++c) {
        const int d0 = 8 * g + 32 * c;
        const _Float16* kr0 = Kp + (size_t)(kb + n) * DD + d0;
        const _Float16* kr1 = Kp + (size_t)(kb + 16 + n) * DD + d0;
        const v16h ka0 = ldfrag(kr0, kr0 + 16);
        const v16h ka1 = ldfrag(kr1, kr1 + 16);
        s0 = wmma16x16x32(ka0, qb[c], s0);
        s1 = wmma16x16x32(ka1, qb[c], s1);
      }

      if (kb + 32 > qbase) {           // diagonal-touching block: causal mask
        const int q = qbase + n;
        #pragma unroll
        for (int r = 0; r < 8; ++r) {
          const int k0 = kb + r + 8 * g;
          if (k0 > q)      s0[r] = MASKVAL;
          if (k0 + 16 > q) s1[r] = MASKVAL;
        }
      }

      // online softmax in log2 domain (exp2 == raw v_exp_f32)
      float bm = s0[0];
      #pragma unroll
      for (int r = 0; r < 8; ++r) { bm = fmaxf(bm, s0[r]); bm = fmaxf(bm, s1[r]); }
      bm = fmaxf(bm, __shfl_xor(bm, 16, 32));
      const float mnew  = fmaxf(m_i, bm);
      const float alpha = exp2f(m_i - mnew);

      v16h pb;                         // lands directly in B-operand layout
      float lsum = 0.0f;
      #pragma unroll
      for (int r = 0; r < 8; ++r) {
        const float p0 = exp2f(s0[r] - mnew);
        const float p1 = exp2f(s1[r] - mnew);
        lsum += p0 + p1;
        pb[r]     = (_Float16)p0;
        pb[r + 8] = (_Float16)p1;
      }
      lsum += __shfl_xor(lsum, 16, 32);
      l_i = l_i * alpha + lsum;
      m_i = mnew;

      #pragma unroll
      for (int t = 0; t < 4; ++t) {
        #pragma unroll
        for (int r = 0; r < 8; ++r) oacc[t][r] *= alpha;
      }

      // O^T += V^T * P^T (V fragments already in registers)
      #pragma unroll
      for (int t = 0; t < 4; ++t) oacc[t] = wmma16x16x32(va[t], pb, oacc[t]);
    }

    const float inv_l = 1.0f / l_i;
    float* orow = Op + (size_t)(qbase + n) * DD;
    #pragma unroll
    for (int t = 0; t < 4; ++t) {
      const int dbase = 16 * t + 8 * g;
      float4 w0, w1;
      w0.x = oacc[t][0] * inv_l; w0.y = oacc[t][1] * inv_l;
      w0.z = oacc[t][2] * inv_l; w0.w = oacc[t][3] * inv_l;
      w1.x = oacc[t][4] * inv_l; w1.y = oacc[t][5] * inv_l;
      w1.z = oacc[t][6] * inv_l; w1.w = oacc[t][7] * inv_l;
      *reinterpret_cast<float4*>(orow + dbase)     = w0;
      *reinterpret_cast<float4*>(orow + dbase + 4) = w1;
    }
  }
}

// ---------------- fallback: self-contained f32-load version ----------------
__global__ __launch_bounds__(256, 1)
void fa_causal_fwd_f32(const float* __restrict__ Q, const float* __restrict__ K,
                       const float* __restrict__ V, float* __restrict__ O) {
  const int lane = threadIdx.x & 31;
  const int wave = threadIdx.x >> 5;
  const int n = lane & 15;
  const int g = lane >> 4;
  const int blocksPerSeq = SS / 128;
  const int bh    = blockIdx.x / blocksPerSeq;
  const int qblk  = blockIdx.x % blocksPerSeq;
  const int qbase = qblk * 128 + wave * 16;

  const float* __restrict__ Qp = Q + (size_t)bh * SS * DD;
  const float* __restrict__ Kp = K + (size_t)bh * SS * DD;
  const float* __restrict__ Vp = V + (size_t)bh * SS * DD;
  float*       __restrict__ Op = O + (size_t)bh * SS * DD;

  v16h qb[2];
  {
    const float* qrow = Qp + (size_t)(qbase + n) * DD;
    #pragma unroll
    for (int c = 0; c < 2; ++c) {
      const int d0 = 8 * g + 32 * c;
      ld8(qb[c], 0, qrow + d0,      QSCALE);
      ld8(qb[c], 8, qrow + d0 + 16, QSCALE);
    }
  }
  v8f oacc[4];
  #pragma unroll
  for (int t = 0; t < 4; ++t) {
    #pragma unroll
    for (int r = 0; r < 8; ++r) oacc[t][r] = 0.0f;
  }
  float m_i = -3.0e38f, l_i = 0.0f;
  const int kend = qbase + 16;
  for (int kb = 0; kb < kend; kb += 32) {
    v8f s0, s1;
    #pragma unroll
    for (int r = 0; r < 8; ++r) { s0[r] = 0.0f; s1[r] = 0.0f; }
    #pragma unroll
    for (int c = 0; c < 2; ++c) {
      const int d0 = 8 * g + 32 * c;
      const float* kr0 = Kp + (size_t)(kb + n) * DD + d0;
      const float* kr1 = Kp + (size_t)(kb + 16 + n) * DD + d0;
      v16h ka0, ka1;
      ld8(ka0, 0, kr0, 1.0f); ld8(ka0, 8, kr0 + 16, 1.0f);
      ld8(ka1, 0, kr1, 1.0f); ld8(ka1, 8, kr1 + 16, 1.0f);
      s0 = wmma16x16x32(ka0, qb[c], s0);
      s1 = wmma16x16x32(ka1, qb[c], s1);
    }
    if (kb + 32 > qbase) {
      const int q = qbase + n;
      #pragma unroll
      for (int r = 0; r < 8; ++r) {
        const int k0 = kb + r + 8 * g;
        if (k0 > q)      s0[r] = MASKVAL;
        if (k0 + 16 > q) s1[r] = MASKVAL;
      }
    }
    float bm = s0[0];
    #pragma unroll
    for (int r = 0; r < 8; ++r) { bm = fmaxf(bm, s0[r]); bm = fmaxf(bm, s1[r]); }
    bm = fmaxf(bm, __shfl_xor(bm, 16, 32));
    const float mnew  = fmaxf(m_i, bm);
    const float alpha = exp2f(m_i - mnew);
    v16h pb;
    float lsum = 0.0f;
    #pragma unroll
    for (int r = 0; r < 8; ++r) {
      const float p0 = exp2f(s0[r] - mnew);
      const float p1 = exp2f(s1[r] - mnew);
      lsum += p0 + p1;
      pb[r]     = (_Float16)p0;
      pb[r + 8] = (_Float16)p1;
    }
    lsum += __shfl_xor(lsum, 16, 32);
    l_i = l_i * alpha + lsum;
    m_i = mnew;
    #pragma unroll
    for (int t = 0; t < 4; ++t) {
      #pragma unroll
      for (int r = 0; r < 8; ++r) oacc[t][r] *= alpha;
    }
    #pragma unroll
    for (int t = 0; t < 4; ++t) {
      const float* vcol = Vp + (size_t)(16 * t + n);
      v16h va;
      #pragma unroll
      for (int e = 0; e < 8; ++e) {
        va[e]     = (_Float16)vcol[(size_t)(kb + 8 * g + e) * DD];
        va[e + 8] = (_Float16)vcol[(size_t)(kb + 16 + 8 * g + e) * DD];
      }
      oacc[t] = wmma16x16x32(va, pb, oacc[t]);
    }
  }
  const float inv_l = 1.0f / l_i;
  float* orow = Op + (size_t)(qbase + n) * DD;
  #pragma unroll
  for (int t = 0; t < 4; ++t) {
    const int dbase = 16 * t + 8 * g;
    float4 w0, w1;
    w0.x = oacc[t][0] * inv_l; w0.y = oacc[t][1] * inv_l;
    w0.z = oacc[t][2] * inv_l; w0.w = oacc[t][3] * inv_l;
    w1.x = oacc[t][4] * inv_l; w1.y = oacc[t][5] * inv_l;
    w1.z = oacc[t][6] * inv_l; w1.w = oacc[t][7] * inv_l;
    *reinterpret_cast<float4*>(orow + dbase)     = w0;
    *reinterpret_cast<float4*>(orow + dbase + 4) = w1;
  }
}

extern "C" void kernel_launch(void* const* d_in, const int* in_sizes, int n_in,
                              void* d_out, int out_size, void* d_ws, size_t ws_size,
                              hipStream_t stream) {
  const float* Q = (const float*)d_in[0];
  const float* K = (const float*)d_in[1];
  const float* V = (const float*)d_in[2];
  // d_in[3] is the causal bool mask; reproduced analytically in-kernel.
  float* O = (float*)d_out;

  const size_t elems = (size_t)BB * HH * SS * DD;   // 8,388,608
  const size_t need  = 2 * elems * sizeof(_Float16);

  dim3 block(256);

  if (ws_size >= need) {
    _Float16* Kh = (_Float16*)d_ws;
    _Float16* Vt = Kh + elems;
    hipLaunchKernelGGL(cvt_f32_f16,      dim3(elems / (256 * 8)),    block, 0, stream, K, Kh);
    hipLaunchKernelGGL(transpose_v_f16,  dim3(BB * HH * (SS / 64)),  block, 0, stream, V, Vt);
    hipLaunchKernelGGL(fa_causal_fwd_h,  dim3(BB * HH * (SS / 256)), block, 0, stream, Q, Kh, Vt, O);
  } else {
    hipLaunchKernelGGL(fa_causal_fwd_f32, dim3(BB * HH * (SS / 128)), block, 0, stream, Q, K, V, O);
  }
}
